// RelationNetwork_83975200571518
// MI455X (gfx1250) — matrix operand order repaired and checked by reading, not compiled, not run on hardware
//
#include <hip/hip_runtime.h>
#include <hip/hip_bf16.h>

typedef __attribute__((ext_vector_type(16))) _Float16 v16h;
typedef __attribute__((ext_vector_type(8)))  float    v8f;

// ---------------- Stage 1: factor layer 1, pre-convert W2 to f16 (transposed) ----
// A[i][k] = x[i]·gW1[0:32,k] + q·gW1[64:96,k] + gb1[k]   (c folded in)
// B[i][k] = x[i]·gW1[32:64,k]
// w2t[n*64+k] = (f16) gW2[k*64+n]   (so one B-fragment dword = contiguous load)
__global__ void rn_stage1(const float* __restrict__ x, const float* __restrict__ q,
                          const float* __restrict__ gW1, const float* __restrict__ gb1,
                          const float* __restrict__ gW2,
                          float* __restrict__ Abuf, float* __restrict__ Bbuf,
                          float* __restrict__ embedAcc, _Float16* __restrict__ w2t)
{
    int t = blockIdx.x * blockDim.x + threadIdx.x;   // 65536 threads
    int i = t >> 6, k = t & 63;
    float a = gb1[k], b = 0.f;
    for (int d = 0; d < 32; ++d) {
        float xv = x[i * 32 + d];
        a += xv   * gW1[d * 64 + k];
        b += xv   * gW1[(32 + d) * 64 + k];
        a += q[d] * gW1[(64 + d) * 64 + k];
    }
    Abuf[t] = a;
    Bbuf[t] = b;
    if (t < 4096) { int n = t >> 6, kk = t & 63; w2t[t] = (_Float16)gW2[kk * 64 + n]; }
    if (t < 64) embedAcc[t] = 0.f;
}

// ---------------- Stage 2: all N^2 pairs via WMMA; accumulate sum of relu(h2) ----
__global__ __launch_bounds__(256) void rn_pairs(
    const float* __restrict__ Abuf, const float* __restrict__ Bbuf,
    const float* __restrict__ gb2, const _Float16* __restrict__ w2t,
    float* __restrict__ embedAcc)
{
    __shared__ float    a_lds[64];
    __shared__ float    block_acc[64];
    __shared__ _Float16 h1[8][16][72];      // per-wave private 16x64 tile (+padding)

    const int tid  = threadIdx.x;
    const int lane = tid & 31;
    const int wv   = tid >> 5;              // wave in block (0..7)
    const int m    = lane & 15;             // row within tile / column within N-tile
    const int hh   = lane >> 4;             // lane half
    const int c0   = hh * 32;               // column half for h1 build
    const int i    = blockIdx.x >> 1;       // object i for this block
    const int half = blockIdx.x & 1;        // j range half

    if (tid < 64) { a_lds[tid] = Abuf[i * 64 + tid]; block_acc[tid] = 0.f; }
    __syncthreads();

    // Loop-invariant W2 B-fragments: 4 N-tiles x 2 K-steps, resident in VGPRs.
    // B k order used here, same as A (16-bit, 32x16): lanes 0-15 carry K=0..15 (2/VGPR), lanes 16-31 K=16..31.
    union HU { v16h h; unsigned u[8]; };
    HU w2f[4][2];
#pragma unroll
    for (int nt = 0; nt < 4; ++nt) {
        const int n = nt * 16 + m;
#pragma unroll
        for (int s = 0; s < 2; ++s)
#pragma unroll
            for (int v = 0; v < 8; ++v) {
                const int k = s * 32 + hh * 16 + 2 * v;
                w2f[nt][s].u[v] = *(const unsigned*)(w2t + n * 64 + k);
            }
    }
    float bias[4];
#pragma unroll
    for (int nt = 0; nt < 4; ++nt) bias[nt] = gb2[nt * 16 + m];

    v8f sum[4];
#pragma unroll
    for (int nt = 0; nt < 4; ++nt)
#pragma unroll
        for (int v = 0; v < 8; ++v) sum[nt][v] = 0.f;

    for (int t4 = 0; t4 < 4; ++t4) {
        const int jt = half * 32 + wv * 4 + t4;     // j-tile (16 j's), i fixed
        const int j  = jt * 16 + m;

        // Build h1 tile: row m = relu(A[i] + B[j]), lane covers 32 columns.
        const float4* brow = (const float4*)(Bbuf + j * 64 + c0);
        const float4* arow = (const float4*)(a_lds + c0);
#pragma unroll
        for (int u = 0; u < 8; ++u) {
            float4 bv = brow[u];
            float4 av = arow[u];
            union { _Float16 h[4]; uint2 u2; } pk;
            pk.h[0] = (_Float16)fmaxf(av.x + bv.x, 0.f);
            pk.h[1] = (_Float16)fmaxf(av.y + bv.y, 0.f);
            pk.h[2] = (_Float16)fmaxf(av.z + bv.z, 0.f);
            pk.h[3] = (_Float16)fmaxf(av.w + bv.w, 0.f);
            *(uint2*)&h1[wv][m][c0 + 4 * u] = pk.u2;
        }
        asm volatile("s_wait_dscnt 0" ::: "memory");  // wave-private LDS: store->load

        v8f c[4];
#pragma unroll
        for (int nt = 0; nt < 4; ++nt)
#pragma unroll
            for (int v = 0; v < 8; ++v) c[nt][v] = bias[nt];  // z2 starts at b2

#pragma unroll
        for (int s = 0; s < 2; ++s) {
            // A k order used here, same as B (16-bit, 16x32): lanes 0-15 K=0..15, lanes 16-31 K=16..31.
            HU af;
#pragma unroll
            for (int v = 0; v < 8; ++v) {
                const int k = s * 32 + hh * 16 + 2 * v;
                af.u[v] = *(const unsigned*)&h1[wv][m][k];
            }
#pragma unroll
            for (int nt = 0; nt < 4; ++nt)
                c[nt] = __builtin_amdgcn_wmma_f32_16x16x32_f16(
                    false, af.h, false, w2f[nt][s].h, (short)0, c[nt], false, false);
        }

        // relu per pair-row, then accumulate the pair-sum in f32 registers.
#pragma unroll
        for (int nt = 0; nt < 4; ++nt)
#pragma unroll
            for (int v = 0; v < 8; ++v)
                sum[nt][v] += fmaxf(c[nt][v], 0.f);
    }

    // Reduce: 8 rows per lane -> cross-half -> LDS -> global.
#pragma unroll
    for (int nt = 0; nt < 4; ++nt) {
        float acc = 0.f;
#pragma unroll
        for (int v = 0; v < 8; ++v) acc += sum[nt][v];
        acc += __shfl_xor(acc, 16, 32);             // wave32 halves
        if (lane < 16) atomicAdd(&block_acc[nt * 16 + lane], acc);
    }
    __syncthreads();
    if (tid < 64) atomicAdd(&embedAcc[tid], block_acc[tid]);
}

// ---------------- Stage 3: subtract the N diagonal (i==i) pairs ------------------
__global__ void rn_diag(const float* __restrict__ Abuf, const float* __restrict__ Bbuf,
                        const float* __restrict__ gW2, const float* __restrict__ gb2,
                        float* __restrict__ embedAcc)
{
    __shared__ float h1d[64];
    const int i = blockIdx.x, t = threadIdx.x;     // 1024 x 64
    h1d[t] = fmaxf(Abuf[i * 64 + t] + Bbuf[i * 64 + t], 0.f);
    __syncthreads();
    float z = gb2[t];
    for (int k = 0; k < 64; ++k) z += h1d[k] * gW2[k * 64 + t];
    atomicAdd(&embedAcc[t], -fmaxf(z, 0.f));
}

// ---------------- Stage 4: layer-3 on the sum, then the f-MLP (single block) -----
__global__ void rn_final(const float* __restrict__ embedAcc,
                         const float* __restrict__ gW3, const float* __restrict__ gb3,
                         const float* __restrict__ fW1, const float* __restrict__ fb1,
                         const float* __restrict__ fW2, const float* __restrict__ fb2,
                         const float* __restrict__ fW3, const float* __restrict__ fb3,
                         float* __restrict__ out)
{
    __shared__ float S[64], E[64], H1[128], H2[128];
    const int t = threadIdx.x;                      // 128 threads
    if (t < 64) S[t] = embedAcc[t];
    __syncthreads();
    if (t < 64) {
        float e = 1047552.0f * gb3[t];              // P = 1024*1023
        for (int k = 0; k < 64; ++k) e += S[k] * gW3[k * 64 + t];
        E[t] = e;
    }
    __syncthreads();
    {
        float h = fb1[t];
        for (int k = 0; k < 64; ++k) h += E[k] * fW1[k * 128 + t];
        H1[t] = fmaxf(h, 0.f);
    }
    __syncthreads();
    {
        float h = fb2[t];
        for (int k = 0; k < 128; ++k) h += H1[k] * fW2[k * 128 + t];
        H2[t] = fmaxf(h, 0.f);
    }
    __syncthreads();
    if (t < 32) {
        float o = fb3[t];
        for (int k = 0; k < 128; ++k) o += H2[k] * fW3[k * 32 + t];
        out[t] = o;
    }
}

extern "C" void kernel_launch(void* const* d_in, const int* in_sizes, int n_in,
                              void* d_out, int out_size, void* d_ws, size_t ws_size,
                              hipStream_t stream) {
    const float* x   = (const float*)d_in[0];
    const float* q   = (const float*)d_in[1];
    const float* gW1 = (const float*)d_in[2];
    const float* gb1 = (const float*)d_in[3];
    const float* gW2 = (const float*)d_in[4];
    const float* gb2 = (const float*)d_in[5];
    const float* gW3 = (const float*)d_in[6];
    const float* gb3 = (const float*)d_in[7];
    const float* fW1 = (const float*)d_in[8];
    const float* fb1 = (const float*)d_in[9];
    const float* fW2 = (const float*)d_in[10];
    const float* fb2 = (const float*)d_in[11];
    const float* fW3 = (const float*)d_in[12];
    const float* fb3 = (const float*)d_in[13];
    float* out = (float*)d_out;

    // Workspace layout (floats): A[65536] | B[65536] | embedAcc[64] | w2t(f16)[4096]
    float*    Abuf     = (float*)d_ws;
    float*    Bbuf     = Abuf + 65536;
    float*    embedAcc = Bbuf + 65536;
    _Float16* w2t      = (_Float16*)(embedAcc + 64);

    rn_stage1<<<256, 256, 0, stream>>>(x, q, gW1, gb1, gW2, Abuf, Bbuf, embedAcc, w2t);
    rn_pairs <<<2048, 256, 0, stream>>>(Abuf, Bbuf, gb2, w2t, embedAcc);
    rn_diag  <<<1024, 64, 0, stream>>>(Abuf, Bbuf, gW2, gb2, embedAcc);
    rn_final <<<1, 128, 0, stream>>>(embedAcc, gW3, gb3, fW1, fb1, fW2, fb2, fW3, fb3, out);
}
